// FieldLstmEncoder_53223234732703
// MI455X (gfx1250) — matrix operand order repaired and checked by reading, not compiled, not run on hardware
//
#include <hip/hip_runtime.h>
#include <hip/hip_bf16.h>

// ---------------------------------------------------------------------------
// FieldLSTM encoder for MI455X (gfx1250, wave32, WMMA f16 16x16x32)
//
// Phase 1 (parallel, hoisted out of time loop):
//   P1[t*B+b][4096] = x_f16 @ W1xT + b1                  (WMMA GEMM, K=512)
//   rdt[t*B+b][1024] = sigmoid(r)*tanh(d), r/d from f@W2T (WMMA GEMM, K=128)
// Phase 2 (512 dependent step kernels):
//   pre = h_prev16 @ W1hT  (WMMA, K=1024, weights L2-resident, LDS-staged)
//   fused LSTM cell epilogue with length masking, h double-buffered in f16
// ---------------------------------------------------------------------------

typedef __attribute__((ext_vector_type(16))) _Float16 v16h;
typedef __attribute__((ext_vector_type(8)))  float    v8f;

#define UNI 512
#define HID 1024
#define FLD 128
#define BB  128
#define SS  512
#define G4  (4*HID)

union ABu { v16h h; uint4 q[2]; };

// A fragment (16x32 f16) from row-major A: per CDNA5 ISA:
// lanes 0-15 hold M=lane, K = {k0..k0+7} in dw0..3 and {k0+16..k0+23} in dw4..7
// lanes 16-31 hold K = {k0+8..k0+15} and {k0+24..k0+31}  -> two b128 loads/lane
__device__ inline v16h load_a16(const _Float16* A, int ld, int m0, int k0, int lane) {
  const int l16 = lane & 15, hi = lane >> 4;
  const _Float16* base = A + (size_t)(m0 + l16) * ld + k0 + hi * 8;
  ABu u;
  u.q[0] = *(const uint4*)(base);
  u.q[1] = *(const uint4*)(base + 16);
  return u.h;
}

// B fragment (32x16 f16) from LDS, pre-packed so each lane's 16 halves are
// contiguous: lanes 0-15 = N, K 0..7,16..23; lanes 16-31 = N, K 8..15,24..31.
__device__ inline v16h load_b_lds(const char* sm, int tile, int kc, int nkch, int lane) {
  const uint4* p = (const uint4*)sm + ((size_t)(tile * nkch + kc) * 32 + lane) * 2;
  ABu u; u.q[0] = p[0]; u.q[1] = p[1];
  return u.h;
}

// Packed-B offset (in halfwords) for element (n = output col, k = K index)
__device__ inline size_t pack_off(int n, int k, int nkch) {
  const int ntile = n >> 4, nn = n & 15;
  const int kchunk = k >> 5, kp = k & 31;
  const int hi = (kp >> 3) & 1;
  const int lane = nn + (hi << 4);
  const int kk = (kp & 7) + ((kp >> 4) << 3); // position within lane = dw*2+half
  return (((size_t)ntile * nkch + kchunk) * 32 + lane) * 16 + kk;
}

__device__ inline float sigm(float x) { return 1.0f / (1.0f + __expf(-x)); }
__device__ inline float tanh_f(float x) {
  x = fminf(fmaxf(x, -15.0f), 15.0f);
  const float e = __expf(2.0f * x);
  return (e - 1.0f) / (e + 1.0f);
}

// ---------------------------------------------------------------------------
// Conversion / packing kernels (run once)
// ---------------------------------------------------------------------------

// x [B,S,UNI] f32 -> time-major f16 [(t*B+b)*UNI + u]
__global__ void k_conv_x(const float* __restrict__ src, _Float16* __restrict__ dst) {
  const size_t total = (size_t)SS * BB * UNI;
  for (size_t idx = (size_t)blockIdx.x * blockDim.x + threadIdx.x; idx < total;
       idx += (size_t)gridDim.x * blockDim.x) {
    const int u = (int)(idx & (UNI - 1));
    const size_t r = idx >> 9;               // /UNI
    const int b = (int)(r & (BB - 1));
    const int t = (int)(r >> 7);             // /BB
    dst[idx] = (_Float16)src[((size_t)b * SS + t) * UNI + u];
  }
}

// field [B,S,FLD] f32 -> time-major f16 [(t*B+b)*FLD + u]
__global__ void k_conv_f(const float* __restrict__ src, _Float16* __restrict__ dst) {
  const size_t total = (size_t)SS * BB * FLD;
  for (size_t idx = (size_t)blockIdx.x * blockDim.x + threadIdx.x; idx < total;
       idx += (size_t)gridDim.x * blockDim.x) {
    const int u = (int)(idx & (FLD - 1));
    const size_t r = idx >> 7;               // /FLD
    const int b = (int)(r & (BB - 1));
    const int t = (int)(r >> 7);
    dst[idx] = (_Float16)src[((size_t)b * SS + t) * FLD + u];
  }
}

// Generic weight packer: W row-major [Nout x src_ld], take K columns starting
// at src_koff, emit f16 in the WMMA B-operand lane layout.
__global__ void k_pack_w(const float* __restrict__ W, _Float16* __restrict__ dst,
                         int kcols, int nkch, int src_ld, int src_koff, size_t total) {
  for (size_t idx = (size_t)blockIdx.x * blockDim.x + threadIdx.x; idx < total;
       idx += (size_t)gridDim.x * blockDim.x) {
    const int n = (int)(idx / kcols);
    const int k = (int)(idx % kcols);
    dst[pack_off(n, k, nkch)] = (_Float16)W[(size_t)n * src_ld + src_koff + k];
  }
}

__global__ void k_init_state(_Float16* __restrict__ h16a,
                             float* __restrict__ h32, float* __restrict__ c32) {
  const int i = blockIdx.x * blockDim.x + threadIdx.x;
  if (i < BB * HID) {
    h16a[i] = (_Float16)0.0f;
    h32[i] = 0.0f;
    c32[i] = 0.0f;
  }
}

// ---------------------------------------------------------------------------
// Phase 1 GEMMs (parallel over all S*B rows)
// ---------------------------------------------------------------------------

// P1[R][4096] = Xf16[R][512] @ W1xT + b1 ; grid (512, 64), block 256, LDS 64KB
__global__ void __launch_bounds__(256) k_proj_in(
    const _Float16* __restrict__ x16, const _Float16* __restrict__ wpack,
    const float* __restrict__ b1, float* __restrict__ p1) {
  extern __shared__ char sm[];
  const int tid = threadIdx.x;
  { // 4 contiguous n-tiles x 16 kchunks = 64KB contiguous slab
    const uint4* src = (const uint4*)(wpack + (size_t)blockIdx.y * 4 * (16 * 32 * 16));
    uint4* dst = (uint4*)sm;
    for (int i = tid; i < 4096; i += 256) dst[i] = src[i];
  }
  __syncthreads();

  const int wave = tid >> 5, lane = tid & 31;
  const int l16 = lane & 15, hi = lane >> 4;
  const int row0 = blockIdx.x * 128 + wave * 16;

  const v8f vz = {0.f, 0.f, 0.f, 0.f, 0.f, 0.f, 0.f, 0.f};
  v8f acc[4] = {vz, vz, vz, vz};

  for (int kc = 0; kc < 16; ++kc) {
    const v16h a = load_a16(x16, UNI, row0, kc * 32, lane);
#pragma unroll
    for (int g = 0; g < 4; ++g) {
      const v16h b = load_b_lds(sm, g, kc, 16, lane);
      acc[g] = __builtin_amdgcn_wmma_f32_16x16x32_f16(false, a, false, b,
                                                      (short)0, acc[g], false, false);
    }
  }

#pragma unroll
  for (int r = 0; r < 8; ++r) {
    const size_t R = (size_t)row0 + hi * 8 + r;
#pragma unroll
    for (int g = 0; g < 4; ++g) {
      const int n = blockIdx.y * 64 + g * 16 + l16;
      p1[R * G4 + n] = acc[g][r] + b1[n];
    }
  }
}

// rdt[R][h] = sigmoid(r)*tanh(d); r/d = Ff16[R] @ W2T + b2 ; grid (512, 64)
__global__ void __launch_bounds__(256) k_proj_field(
    const _Float16* __restrict__ f16v, const _Float16* __restrict__ wpack,
    const float* __restrict__ b2, float* __restrict__ rdt) {
  extern __shared__ char sm[];
  const int tid = threadIdx.x;
  { // two gate-strided n-tiles (r at ntile y, d at ntile 64+y), 4KB each
    const uint4* srcR = (const uint4*)(wpack + (size_t)blockIdx.y * (4 * 32 * 16));
    const uint4* srcD = (const uint4*)(wpack + (size_t)(64 + blockIdx.y) * (4 * 32 * 16));
    uint4* dst = (uint4*)sm;
    if (tid < 256) {
      dst[tid] = srcR[tid];
      dst[256 + tid] = srcD[tid];
    }
  }
  __syncthreads();

  const int wave = tid >> 5, lane = tid & 31;
  const int l16 = lane & 15, hi = lane >> 4;
  const int row0 = blockIdx.x * 128 + wave * 16;

  const v8f vz = {0.f, 0.f, 0.f, 0.f, 0.f, 0.f, 0.f, 0.f};
  v8f accR = vz, accD = vz;

  for (int kc = 0; kc < 4; ++kc) {
    const v16h a = load_a16(f16v, FLD, row0, kc * 32, lane);
    const v16h br = load_b_lds(sm, 0, kc, 4, lane);
    accR = __builtin_amdgcn_wmma_f32_16x16x32_f16(false, a, false, br,
                                                  (short)0, accR, false, false);
    const v16h bd = load_b_lds(sm, 1, kc, 4, lane);
    accD = __builtin_amdgcn_wmma_f32_16x16x32_f16(false, a, false, bd,
                                                  (short)0, accD, false, false);
  }

  const int h = blockIdx.y * 16 + l16;
#pragma unroll
  for (int r = 0; r < 8; ++r) {
    const size_t R = (size_t)row0 + hi * 8 + r;
    const float rr = accR[r] + b2[h];
    const float dd = accD[r] + b2[HID + h];
    rdt[R * HID + h] = sigm(rr) * tanh_f(dd);
  }
}

// ---------------------------------------------------------------------------
// Phase 2: fused recurrent step (launched S times, stream-ordered)
// grid 64 blocks x 256 threads; block owns hidden cols [16*bx,16*bx+16) for
// ALL four gates -> full cell update fused in the epilogue. LDS 128KB.
// ---------------------------------------------------------------------------
__global__ void __launch_bounds__(256) k_lstm_step(
    const _Float16* __restrict__ hprev, _Float16* __restrict__ hnew,
    float* __restrict__ h32, float* __restrict__ c32,
    const _Float16* __restrict__ w1h_pack,
    const float* __restrict__ p1, const float* __restrict__ rdt,
    const int* __restrict__ len, float* __restrict__ out, int t) {
  extern __shared__ char sm[];
  const int tid = threadIdx.x;
  // Stage the 4 gate-strided weight tiles (each 32KB) -> 128KB LDS
  for (int g = 0; g < 4; ++g) {
    const int ntile = g * 64 + blockIdx.x;
    const uint4* src = (const uint4*)(w1h_pack + (size_t)ntile * (32 * 32 * 16));
    uint4* dst = (uint4*)sm + (size_t)g * 2048;
    for (int i = tid; i < 2048; i += 256) dst[i] = src[i];
  }
  __syncthreads();

  const int wave = tid >> 5, lane = tid & 31;
  const int l16 = lane & 15, hi = lane >> 4;
  const int m0 = wave * 16;                 // batch rows for this wave
  const size_t R0 = (size_t)t * BB;

  __builtin_prefetch(&p1[(R0 + m0) * G4], 0, 1);          // global_prefetch_b8
  __builtin_prefetch(&rdt[(R0 + m0) * HID], 0, 1);

  const v8f vz = {0.f, 0.f, 0.f, 0.f, 0.f, 0.f, 0.f, 0.f};
  v8f acc[4] = {vz, vz, vz, vz};

  for (int kc = 0; kc < 32; ++kc) {
    const v16h a = load_a16(hprev, HID, m0, kc * 32, lane);
#pragma unroll
    for (int g = 0; g < 4; ++g) {
      const v16h b = load_b_lds(sm, g, kc, 32, lane);
      acc[g] = __builtin_amdgcn_wmma_f32_16x16x32_f16(false, a, false, b,
                                                      (short)0, acc[g], false, false);
    }
  }

  const int h = blockIdx.x * 16 + l16;
#pragma unroll
  for (int r = 0; r < 8; ++r) {
    const int b = m0 + hi * 8 + r;
    const size_t R = R0 + b;
    const float pi = acc[0][r] + p1[R * G4 + 0 * HID + h];
    const float pj = acc[1][r] + p1[R * G4 + 1 * HID + h];
    const float pf = acc[2][r] + p1[R * G4 + 2 * HID + h];
    const float po = acc[3][r] + p1[R * G4 + 3 * HID + h];
    const size_t si = (size_t)b * HID + h;
    const float cn = sigm(pf + 1.0f) * c32[si] + sigm(pi) * tanh_f(pj) + rdt[R * HID + h];
    const float hv = sigm(po) * tanh_f(cn);
    const size_t oidx = ((size_t)b * SS + t) * (size_t)HID + h;
    if (t >= len[b]) {                       // finished sequence: hold state
      out[oidx] = 0.0f;
      hnew[si] = hprev[si];
    } else {
      out[oidx] = hv;
      hnew[si] = (_Float16)hv;
      h32[si] = hv;
      c32[si] = cn;
    }
  }
}

__global__ void k_finalize(const float* __restrict__ h32,
                           const float* __restrict__ c32,
                           float* __restrict__ out_tail) {
  const int i = blockIdx.x * blockDim.x + threadIdx.x;
  if (i < BB * HID) {
    out_tail[i] = h32[i];
    out_tail[BB * HID + i] = c32[i];
  }
}

// ---------------------------------------------------------------------------
// Host launcher
// ---------------------------------------------------------------------------
extern "C" void kernel_launch(void* const* d_in, const int* in_sizes, int n_in,
                              void* d_out, int out_size, void* d_ws, size_t ws_size,
                              hipStream_t stream) {
  (void)in_sizes; (void)n_in; (void)out_size; (void)ws_size;

  const float* x   = (const float*)d_in[0];   // [B,S,UNI]
  const float* fld = (const float*)d_in[1];   // [B,S,FLD]
  const int*   len = (const int*)d_in[2];     // [B]
  const float* W1  = (const float*)d_in[3];   // [4H, UNI+HID]
  const float* b1  = (const float*)d_in[4];   // [4H]
  const float* W2  = (const float*)d_in[5];   // [2H, FLD]
  const float* b2  = (const float*)d_in[6];   // [2H]
  float* out = (float*)d_out;

  // workspace carve-up (256B aligned)
  char* ws = (char*)d_ws;
  size_t off = 0;
  auto take = [&](size_t bytes) {
    char* p = ws + off;
    off += (bytes + 255) & ~(size_t)255;
    return p;
  };
  _Float16* x16   = (_Float16*)take((size_t)SS * BB * UNI * 2);   // 64 MB
  _Float16* f16v  = (_Float16*)take((size_t)SS * BB * FLD * 2);   // 16 MB
  _Float16* pw1x  = (_Float16*)take((size_t)G4 * UNI * 2);        // 4 MB
  _Float16* pw1h  = (_Float16*)take((size_t)G4 * HID * 2);        // 8 MB
  _Float16* pw2   = (_Float16*)take((size_t)2 * HID * FLD * 2);   // 512 KB
  float*    p1    = (float*)take((size_t)SS * BB * G4 * 4);       // 1 GB
  float*    rdt   = (float*)take((size_t)SS * BB * HID * 4);      // 256 MB
  _Float16* h16   = (_Float16*)take((size_t)2 * BB * HID * 2);    // double buffer
  float*    h32   = (float*)take((size_t)BB * HID * 4);
  float*    c32   = (float*)take((size_t)BB * HID * 4);

  // --- once-per-call setup (deterministic; no caching allowed) ---
  k_conv_x<<<8192, 256, 0, stream>>>(x, x16);
  k_conv_f<<<4096, 256, 0, stream>>>(fld, f16v);
  k_pack_w<<<4096, 256, 0, stream>>>(W1, pw1x, UNI, 16, UNI + HID, 0,
                                     (size_t)G4 * UNI);
  k_pack_w<<<8192, 256, 0, stream>>>(W1, pw1h, HID, 32, UNI + HID, UNI,
                                     (size_t)G4 * HID);
  k_pack_w<<<1024, 256, 0, stream>>>(W2, pw2, FLD, 4, FLD, 0,
                                     (size_t)2 * HID * FLD);
  k_init_state<<<(BB * HID + 255) / 256, 256, 0, stream>>>(h16, h32, c32);

  // --- phase 1: hoisted input/field projections (pure WMMA, parallel) ---
  k_proj_in<<<dim3(SS * BB / 128, G4 / 64), 256, 65536, stream>>>(x16, pw1x, b1, p1);
  k_proj_field<<<dim3(SS * BB / 128, HID / 16), 256, 8192, stream>>>(f16v, pw2, b2, rdt);

  // --- phase 2: 512 dependent fused recurrent steps ---
  const size_t NH = (size_t)BB * HID;
  for (int t = 0; t < SS; ++t) {
    const _Float16* hA = h16 + (size_t)(t & 1) * NH;
    _Float16* hB = h16 + (size_t)((t + 1) & 1) * NH;
    k_lstm_step<<<HID / 16, 256, 131072, stream>>>(hA, hB, h32, c32, pw1h,
                                                   p1, rdt, len, out, t);
  }

  // --- final (h, c) tail of d_out ---
  float* tail = out + (size_t)BB * SS * HID;
  k_finalize<<<(BB * HID + 255) / 256, 256, 0, stream>>>(h32, c32, tail);
}
